// PoseidonMoE_37546604102323
// MI455X (gfx1250) — matrix operand order repaired and checked by reading, not compiled, not run on hardware
//
#include <hip/hip_runtime.h>
#include <hip/hip_bf16.h>

// ---------------------------------------------------------------------------
// PoseidonMoE for MI455X (gfx1250). All GEMM-shaped math via
// v_wmma_f32_16x16x32_f16 with f16 operands staged once per launch
// (fp32 -> f16 pre-cast), f32 accumulate, vectorized b128 fragment loads.
// B=8, N=4096, D=256, T=768, E=4, K=2, GS=64, WS=8, NH=4, MODES=16, HID=1024
// ---------------------------------------------------------------------------

typedef __attribute__((ext_vector_type(16))) _Float16 v16h;
typedef __attribute__((ext_vector_type(8)))  _Float16 v8h;
typedef __attribute__((ext_vector_type(8)))  float    v8f;

#define BB   8
#define NN   4096
#define DD   256
#define TT   768
#define GSZ  64
#define MODES 16
#define HID  1024
#define MTOK (BB * NN)          // 32768 rows

__device__ __forceinline__ float gelu_f(float v) {
    return 0.5f * v * (1.0f + erff(v * 0.70710678118654752f));
}

// --- f16 fragment loaders (contiguous k order, same for A and B) -----------------------
// A (16x32): lane holds row M=lane&15; halves at K = hi*16..+16 (contiguous)
__device__ __forceinline__ v16h load_a_frag(const _Float16* arow) {
    const int hi = (threadIdx.x >> 4) & 1;
    return *(const v16h*)(arow + hi * 16);
}
// B (32x16): lane holds col N=lane&15; halves at K = hi*16..+16 (contiguous)
__device__ __forceinline__ v16h load_b_frag(const _Float16* brow) {
    const int hi = (threadIdx.x >> 4) & 1;
    return *(const v16h*)(brow + hi * 16);
}

__device__ __forceinline__ void wmma_step_h(const _Float16* arow, const _Float16* brow,
                                            v8f& acc) {
    v16h a = load_a_frag(arow);
    v16h b = load_b_frag(brow);
    acc = __builtin_amdgcn_wmma_f32_16x16x32_f16(false, a, false, b, (short)0, acc,
                                                 false, false);
}

// Strided fp32 variant for the FNO spectral contraction (A strided over
// channel i, B strided over i inside [i,o,x,y] weights). bsgn=-1 => -Ai*Bi.
__device__ __forceinline__ void wmma_step_strided(const float* a, size_t astr,
                                                  const float* b, size_t bstr,
                                                  float bsgn, v8f& acc) {
    const int hi = (threadIdx.x >> 4) & 1;
    v16h av, bv;
#pragma unroll
    for (int e = 0; e < 16; ++e) av[e] = (_Float16)a[(size_t)(hi * 16 + e) * astr];
#pragma unroll
    for (int e = 0; e < 16; ++e) bv[e] = (_Float16)(bsgn * b[(size_t)(hi * 16 + e) * bstr]);
    acc = __builtin_amdgcn_wmma_f32_16x16x32_f16(false, av, false, bv, (short)0, acc,
                                                 false, false);
}

// ---------------------------------------------------------------------------
// fp32 -> f16 cast (weights / activations staged once per launch)
// ---------------------------------------------------------------------------
__global__ void cvt_kernel(const float* __restrict__ in, _Float16* __restrict__ out,
                           size_t n) {
    const size_t i = (size_t)blockIdx.x * 256 + threadIdx.x;
    if (i < n) out[i] = (_Float16)in[i];
}

// ---------------------------------------------------------------------------
// WMMA GEMM: out = [ACT]( A[M,K](f16) @ W[Nout,K](f16)^T + bias ) (+residual)
// COMB: out(f32) += combine[row/4096*4 + expert] * val (expert mixing).
// Each wave owns a 16x64 strip (one A fragment feeds 4 independent WMMAs).
// ---------------------------------------------------------------------------
template <int ACT, int OUTF16, int RES, int COMB>
__global__ __launch_bounds__(256) void gemm_kernel(
    const _Float16* __restrict__ A, int lda, const _Float16* __restrict__ W, int ldw,
    const float* __restrict__ bias, void* __restrict__ out, int ldo,
    const float* __restrict__ residual, int ldr,
    const float* __restrict__ combine, int expert, int M, int Kdim, int Nout) {
    const int wave = threadIdx.x >> 5;
    const int lane = threadIdx.x & 31;
    const int r15 = lane & 15;
    const int hi  = lane >> 4;
    const int nq  = Nout >> 6;                  // quads of 16-wide N tiles
    long t = (long)blockIdx.x * 8 + wave;
    long total = (long)(M >> 4) * nq;
    if (t >= total) return;                     // uniform per wave
    const int mt = (int)(t / nq);
    const int q  = (int)(t % nq);
    const _Float16* arow  = A + (size_t)(mt * 16 + r15) * lda;
    const _Float16* wrow0 = W + (size_t)(q * 64 + r15) * ldw;
    v8f acc[4] = {};
    for (int k0 = 0; k0 < Kdim; k0 += 32) {
        __builtin_prefetch(arow + k0 + 128, 0, 3);    // global_prefetch_b8
        v16h a = load_a_frag(arow + k0);
#pragma unroll
        for (int j = 0; j < 4; ++j) {
            v16h b = load_b_frag(wrow0 + (size_t)j * 16 * ldw + k0);
            acc[j] = __builtin_amdgcn_wmma_f32_16x16x32_f16(false, a, false, b,
                                                            (short)0, acc[j],
                                                            false, false);
        }
    }
#pragma unroll
    for (int j = 0; j < 4; ++j) {
        const int col = q * 64 + j * 16 + r15;
        const float bv = bias[col];
#pragma unroll
        for (int r = 0; r < 8; ++r) {
            const int row = mt * 16 + r + 8 * hi;
            float v = acc[j][r] + bv;
            if (ACT == 1) v = gelu_f(v);
            if (RES) v += residual[(size_t)row * ldr + col];
            if (COMB) {
                const float s = combine[(row >> 12) * 4 + expert];   // row/4096 = b
                ((float*)out)[(size_t)row * ldo + col] += s * v;
            } else if (OUTF16) {
                ((_Float16*)out)[(size_t)row * ldo + col] = (_Float16)v;
            } else {
                ((float*)out)[(size_t)row * ldo + col] = v;
            }
        }
    }
}

// ---------------------------------------------------------------------------
// Window attention: one block (4 waves) per (window, head). WMMA for Q*K^T and
// P*V; softmax across 16-lane groups matching the C-fragment layout.
// ---------------------------------------------------------------------------
__global__ __launch_bounds__(128) void win_attn_kernel(const _Float16* __restrict__ qkv,
                                                       _Float16* __restrict__ attn_out) {
    __shared__ _Float16 lds_vt[64 * 64];   // V^T [j][l]
    __shared__ _Float16 lds_p[64 * 64];    // probabilities [l][k]
    const int win  = blockIdx.x >> 2;      // 0..511
    const int head = blockIdx.x & 3;
    const int b  = win >> 6;
    const int w_ = win & 63;
    const int wy = w_ >> 3, wx = w_ & 7;
    const int tid = threadIdx.x;

    for (int e = tid; e < 64 * 64; e += 128) {   // stage V^T
        const int j = e >> 6, l = e & 63;
        const size_t tok = (size_t)b * NN + (size_t)(wy * 8 + (l >> 3)) * 64 + wx * 8 + (l & 7);
        lds_vt[j * 64 + l] = qkv[tok * 768 + 512 + head * 64 + j];
    }
    __syncthreads();

    const int wave = tid >> 5, lane = tid & 31, r15 = lane & 15, hi = lane >> 4;
    const int lq = wave * 16 + r15;
    const size_t tokq = (size_t)b * NN + (size_t)(wy * 8 + (lq >> 3)) * 64 + wx * 8 + (lq & 7);
    const _Float16* qrow = qkv + tokq * 768 + head * 64;

    v8f accS[4] = {};
#pragma unroll
    for (int nt = 0; nt < 4; ++nt) {
        const int lk = nt * 16 + r15;
        const size_t tokk = (size_t)b * NN + (size_t)(wy * 8 + (lk >> 3)) * 64 + wx * 8 + (lk & 7);
        const _Float16* krow = qkv + tokk * 768 + 256 + head * 64;
        wmma_step_h(qrow + 0,  krow + 0,  accS[nt]);
        wmma_step_h(qrow + 32, krow + 32, accS[nt]);
    }
    const float scale = 0.125f;               // 1/sqrt(64)
#pragma unroll
    for (int r = 0; r < 8; ++r) {
        float mx = -3.0e38f;
#pragma unroll
        for (int t = 0; t < 4; ++t) { float v = accS[t][r] * scale; accS[t][r] = v; mx = fmaxf(mx, v); }
        for (int off = 1; off < 16; off <<= 1) mx = fmaxf(mx, __shfl_xor(mx, off, 32));
        float sum = 0.f;
#pragma unroll
        for (int t = 0; t < 4; ++t) { float p = __expf(accS[t][r] - mx); accS[t][r] = p; sum += p; }
        for (int off = 1; off < 16; off <<= 1) sum += __shfl_xor(sum, off, 32);
        const float inv = 1.0f / sum;
        const int lrow = wave * 16 + r + 8 * hi;
#pragma unroll
        for (int t = 0; t < 4; ++t)
            lds_p[lrow * 64 + t * 16 + r15] = (_Float16)(accS[t][r] * inv);
    }
    __syncthreads();

    const _Float16* prow = lds_p + (wave * 16 + r15) * 64;
#pragma unroll
    for (int nt = 0; nt < 4; ++nt) {
        v8f acc = {};
        const _Float16* vrow = lds_vt + (nt * 16 + r15) * 64;
        wmma_step_h(prow + 0,  vrow + 0,  acc);
        wmma_step_h(prow + 32, vrow + 32, acc);
#pragma unroll
        for (int r = 0; r < 8; ++r) {
            const int l = wave * 16 + r + 8 * hi;
            const size_t tok = (size_t)b * NN + (size_t)(wy * 8 + (l >> 3)) * 64 + wx * 8 + (l & 7);
            attn_out[tok * 256 + head * 64 + nt * 16 + r15] = (_Float16)acc[r];
        }
    }
}

// ---------------------------------------------------------------------------
// LayerNorm over D=256, one token per 256-thread block, f16 output (GEMM A).
// ---------------------------------------------------------------------------
__global__ __launch_bounds__(256) void ln_kernel(const float* __restrict__ in,
                                                 const float* __restrict__ g,
                                                 const float* __restrict__ bta,
                                                 _Float16* __restrict__ out) {
    __shared__ float red[256];
    const size_t tok = blockIdx.x;
    const int d = threadIdx.x;
    const float v = in[tok * 256 + d];
    red[d] = v; __syncthreads();
    for (int s = 128; s > 0; s >>= 1) { if (d < s) red[d] += red[d + s]; __syncthreads(); }
    const float mean = red[0] * (1.0f / 256.0f);
    __syncthreads();
    const float c = v - mean;
    red[d] = c * c; __syncthreads();
    for (int s = 128; s > 0; s >>= 1) { if (d < s) red[d] += red[d + s]; __syncthreads(); }
    const float var = red[0] * (1.0f / 256.0f);
    out[tok * 256 + d] = (_Float16)(c * rsqrtf(var + 1e-5f) * g[d] + bta[d]);
}

// ---------------------------------------------------------------------------
// Gating: token mean + text concat, 2-layer MLP, top-2 softmax -> combine, w
// ---------------------------------------------------------------------------
__global__ void mean_concat_kernel(const float* __restrict__ x,
                                   const float* __restrict__ text,
                                   float* __restrict__ feat) {
    const int i = blockIdx.x * 256 + threadIdx.x;
    if (i >= BB * (DD + TT)) return;
    const int b = i >> 10, j = i & 1023;
    if (j < DD) {
        float s = 0.f;
        const float* p = x + (size_t)b * NN * DD + j;
        for (int n = 0; n < NN; ++n) s += p[(size_t)n * DD];
        feat[i] = s * (1.0f / (float)NN);
    } else {
        feat[i] = text[b * TT + (j - DD)];
    }
}

__global__ void gating_kernel(const float* __restrict__ feat,
                              const float* __restrict__ g1w, const float* __restrict__ g1b,
                              const float* __restrict__ g2w, const float* __restrict__ g2b,
                              float* __restrict__ combine, float* __restrict__ w_out) {
    __shared__ float sh[BB * DD];
    __shared__ float lg[BB * 4];
    const int t = threadIdx.x;
    for (int e = t; e < BB * DD; e += 256) {
        const int b = e >> 8, d = e & 255;
        const float* f = feat + b * (DD + TT);
        const float* w = g1w + (size_t)d * (DD + TT);
        float s = g1b[d];
        for (int k = 0; k < DD + TT; ++k) s += f[k] * w[k];
        sh[e] = fmaxf(s, 0.f);
    }
    __syncthreads();
    if (t < 32) {
        const int b = t >> 2, e = t & 3;
        const float* h = sh + b * DD;
        const float* w = g2w + e * DD;
        float s = g2b[e];
        for (int k = 0; k < DD; ++k) s += h[k] * w[k];
        lg[t] = s;
    }
    __syncthreads();
    if (t < BB) {
        const int b = t;
        float v[4];
        for (int e = 0; e < 4; ++e) v[e] = lg[b * 4 + e];
        int i0 = 0;
        for (int e = 1; e < 4; ++e) if (v[e] > v[i0]) i0 = e;
        int i1 = -1;
        for (int e = 0; e < 4; ++e) { if (e == i0) continue; if (i1 < 0 || v[e] > v[i1]) i1 = e; }
        const float e1 = __expf(v[i1] - v[i0]);
        const float inv = 1.0f / (1.0f + e1);
        const float w0 = inv, w1 = e1 * inv;
        for (int e = 0; e < 4; ++e) combine[b * 4 + e] = 0.f;
        combine[b * 4 + i0] = w0;
        combine[b * 4 + i1] = w1;
        w_out[b * 2 + 0] = w0;
        w_out[b * 2 + 1] = w1;
    }
}

__global__ void zero_kernel(float* __restrict__ p, size_t n) {
    const size_t i = (size_t)blockIdx.x * 256 + threadIdx.x;
    if (i < n) p[i] = 0.f;
}

// ---------------------------------------------------------------------------
// FNO: factored 64-point DFTs (VALU, tiny cost) + WMMA spectral contraction.
// ---------------------------------------------------------------------------
__global__ __launch_bounds__(256) void fno_fft1(const float* __restrict__ x,
                                                float* __restrict__ tr, float* __restrict__ ti) {
    __shared__ float ct[64], st[64];
    if (threadIdx.x < 64) {
        const float a = -6.283185307f * (float)threadIdx.x / 64.f;
        ct[threadIdx.x] = __cosf(a); st[threadIdx.x] = __sinf(a);
    }
    __syncthreads();
    const int idx = blockIdx.x * 256 + threadIdx.x;          // [b,i,h,ky]
    if (idx >= BB * DD * GSZ * MODES) return;
    const int ky = idx & 15, h = (idx >> 4) & 63, i = (idx >> 10) & 255, b = idx >> 18;
    const float* p = x + ((size_t)b * NN + (size_t)h * 64) * DD + i;
    float sr = 0.f, si = 0.f;
    for (int w = 0; w < 64; ++w) {
        const float v = p[(size_t)w * DD];
        const int ph = (ky * w) & 63;
        sr += v * ct[ph]; si += v * st[ph];
    }
    tr[idx] = sr; ti[idx] = si;
}

__global__ __launch_bounds__(256) void fno_fft2(const float* __restrict__ tr,
                                                const float* __restrict__ ti,
                                                float* __restrict__ xr, float* __restrict__ xi) {
    __shared__ float ct[64], st[64];
    if (threadIdx.x < 64) {
        const float a = -6.283185307f * (float)threadIdx.x / 64.f;
        ct[threadIdx.x] = __cosf(a); st[threadIdx.x] = __sinf(a);
    }
    __syncthreads();
    const int idx = blockIdx.x * 256 + threadIdx.x;          // [b,i,r,ky]
    if (idx >= BB * DD * 32 * MODES) return;
    const int ky = idx & 15, r = (idx >> 4) & 31, i = (idx >> 9) & 255, b = idx >> 17;
    const int kx = (r < 16) ? r : r + 32;
    const float* pr = tr + ((size_t)(b * DD + i) * 64) * 16 + ky;
    const float* pi = ti + ((size_t)(b * DD + i) * 64) * 16 + ky;
    float sr = 0.f, si = 0.f;
    for (int h = 0; h < 64; ++h) {
        const int ph = (kx * h) & 63;
        const float c = ct[ph], s = st[ph];
        const float ar = pr[(size_t)h * 16], ai = pi[(size_t)h * 16];
        sr += ar * c - ai * s;
        si += ar * s + ai * c;
    }
    xr[idx] = sr; xi[idx] = si;
}

// out_ft[b,o,rm,ky] = sum_i xft[b,i,rm,ky] * w[i,o,rm',ky]  (complex, WMMA)
__global__ __launch_bounds__(256) void fno_spectral(
    const float* __restrict__ xr, const float* __restrict__ xi,
    const float* __restrict__ w1r, const float* __restrict__ w1i,
    const float* __restrict__ w2r, const float* __restrict__ w2i,
    float* __restrict__ outr, float* __restrict__ outi) {
    const int wave = threadIdx.x >> 5, lane = threadIdx.x & 31;
    const int r15 = lane & 15, hi = lane >> 4;
    const int task = blockIdx.x * 8 + wave;    // 0..8191
    const int ot = task & 15;                  // output-channel tile
    const int mode = task >> 4;                // 0..511
    const int ky = mode & 15;
    const int rm = mode >> 4;                  // 0..31 (0..15 -> w1, 16..31 -> w2)
    const float* wr = (rm < 16) ? w1r : w2r;
    const float* wi = (rm < 16) ? w1i : w2i;
    const int xm = (rm < 16) ? rm : (rm - 16);
    const int brow = r15 & 7;                  // A row = batch (valid 0..7)
    const size_t abase = (size_t)brow * (DD * 32 * 16) + (size_t)rm * 16 + ky;
    const float* ar = xr + abase;
    const float* ai = xi + abase;
    const int o = ot * 16 + r15;               // B col = output channel
    const size_t bbase = (size_t)o * 256 + (size_t)xm * 16 + ky;
    const float* brp = wr + bbase;
    const float* bip = wi + bbase;
    v8f aR = {}, aI = {};
    for (int k0 = 0; k0 < DD; k0 += 32) {
        const float* arp = ar + (size_t)k0 * 512;
        const float* aip = ai + (size_t)k0 * 512;
        const float* brk = brp + (size_t)k0 * 65536;
        const float* bik = bip + (size_t)k0 * 65536;
        wmma_step_strided(arp, 512, brk, 65536,  1.0f, aR);  // + Ar*Br
        wmma_step_strided(aip, 512, bik, 65536, -1.0f, aR);  // - Ai*Bi
        wmma_step_strided(arp, 512, bik, 65536,  1.0f, aI);  // + Ar*Bi
        wmma_step_strided(aip, 512, brk, 65536,  1.0f, aI);  // + Ai*Br
    }
    if (hi == 0) {
#pragma unroll
        for (int r = 0; r < 8; ++r) {          // D row r = batch
            const size_t oidx = (size_t)r * (DD * 32 * 16) + (size_t)o * 512 + rm * 16 + ky;
            outr[oidx] = aR[r];
            outi[oidx] = aI[r];
        }
    }
}

__global__ __launch_bounds__(256) void fno_ifft1(const float* __restrict__ fr,
                                                 const float* __restrict__ fi,
                                                 float* __restrict__ ur, float* __restrict__ ui) {
    __shared__ float ct[64], st[64];
    if (threadIdx.x < 64) {
        const float a = 6.283185307f * (float)threadIdx.x / 64.f;   // +theta
        ct[threadIdx.x] = __cosf(a); st[threadIdx.x] = __sinf(a);
    }
    __syncthreads();
    const int idx = blockIdx.x * 256 + threadIdx.x;          // [b,o,h,ky]
    if (idx >= BB * DD * GSZ * MODES) return;
    const int ky = idx & 15, h = (idx >> 4) & 63, o = (idx >> 10) & 255, b = idx >> 18;
    const float* pr = fr + (size_t)b * (DD * 32 * 16) + (size_t)o * 512 + ky;
    const float* pi = fi + (size_t)b * (DD * 32 * 16) + (size_t)o * 512 + ky;
    float sr = 0.f, si = 0.f;
    for (int r = 0; r < 32; ++r) {
        const int kx = (r < 16) ? r : r + 32;
        const int ph = (kx * h) & 63;
        const float c = ct[ph], s = st[ph];
        const float a0 = pr[r * 16], a1 = pi[r * 16];
        sr += a0 * c - a1 * s;
        si += a0 * s + a1 * c;
    }
    ur[idx] = sr; ui[idx] = si;
}

// ifft over ky + hermitian doubling + 1x1-conv add + gelu + residual + combine
__global__ __launch_bounds__(256) void fno_final(const float* __restrict__ ur,
                                                 const float* __restrict__ ui,
                                                 const float* __restrict__ x2,
                                                 const float* __restrict__ x,
                                                 const float* __restrict__ combine,
                                                 float* __restrict__ outp) {
    __shared__ float ct[64], st[64];
    if (threadIdx.x < 64) {
        const float a = 6.283185307f * (float)threadIdx.x / 64.f;
        ct[threadIdx.x] = __cosf(a); st[threadIdx.x] = __sinf(a);
    }
    __syncthreads();
    const size_t idx = (size_t)blockIdx.x * 256 + threadIdx.x;   // [b,o,h,w]
    const int w = (int)(idx & 63), h = (int)((idx >> 6) & 63);
    const int o = (int)((idx >> 12) & 255), b = (int)(idx >> 20);
    const float* pr = ur + ((size_t)(b * DD + o) * 64 + h) * 16;
    const float* pi = ui + ((size_t)(b * DD + o) * 64 + h) * 16;
    float acc = pr[0];
    for (int ky = 1; ky < 16; ++ky) {
        const int ph = (ky * w) & 63;
        acc += 2.0f * (pr[ky] * ct[ph] - pi[ky] * st[ph]);
    }
    const float x1 = acc * (1.0f / 4096.0f);
    const size_t tok = (size_t)b * NN + (size_t)h * 64 + w;
    const float v = gelu_f(x1 + x2[tok * DD + o]) + x[tok * DD + o];
    outp[tok * DD + o] += combine[b * 4 + 0] * v;
}

// ---------------------------------------------------------------------------
// Host launch sequence
// ---------------------------------------------------------------------------
extern "C" void kernel_launch(void* const* d_in, const int* in_sizes, int n_in,
                              void* d_out, int out_size, void* d_ws, size_t ws_size,
                              hipStream_t stream) {
    (void)in_sizes; (void)n_in; (void)out_size; (void)ws_size;
    const float* x    = (const float*)d_in[0];
    const float* text = (const float*)d_in[1];
    const float* g1w  = (const float*)d_in[2];
    const float* g1b  = (const float*)d_in[3];
    const float* g2w  = (const float*)d_in[4];
    const float* g2b  = (const float*)d_in[5];
    const float* w1r  = (const float*)d_in[6];
    const float* w1i  = (const float*)d_in[7];
    const float* w2r  = (const float*)d_in[8];
    const float* w2i  = (const float*)d_in[9];
    const float* cw   = (const float*)d_in[10];
    const float* cb   = (const float*)d_in[11];
    const float* m1w1 = (const float*)d_in[12];
    const float* m1b1 = (const float*)d_in[13];
    const float* m1w2 = (const float*)d_in[14];
    const float* m1b2 = (const float*)d_in[15];
    const float* ln1w = (const float*)d_in[16];
    const float* ln1b = (const float*)d_in[17];
    const float* qkvw = (const float*)d_in[18];
    const float* qkvb = (const float*)d_in[19];
    const float* ow   = (const float*)d_in[20];
    const float* ob   = (const float*)d_in[21];
    const float* ln2w = (const float*)d_in[22];
    const float* ln2b = (const float*)d_in[23];
    const float* amw1 = (const float*)d_in[24];
    const float* amb1 = (const float*)d_in[25];
    const float* amw2 = (const float*)d_in[26];
    const float* amb2 = (const float*)d_in[27];
    const float* m2w1 = (const float*)d_in[28];
    const float* m2b1 = (const float*)d_in[29];
    const float* m2w2 = (const float*)d_in[30];
    const float* m2b2 = (const float*)d_in[31];

    float* out  = (float*)d_out;
    float* wout = out + (size_t)MTOK * DD;       // w [B,K] tail

    char* ws = (char*)d_ws;
    size_t off = 0;
    auto alloc = [&](size_t bytes) -> char* {
        char* p = ws + off;
        off = (off + bytes + 255) & ~(size_t)255;
        return p;
    };
    float*    combine = (float*)alloc(BB * 4 * sizeof(float));
    float*    feat    = (float*)alloc((size_t)BB * (DD + TT) * sizeof(float));
    char*     buf1    = alloc((size_t)MTOK * HID * sizeof(_Float16));  // 64MB
    float*    buf2    = (float*)alloc((size_t)MTOK * DD * sizeof(float));
    _Float16* buf3    = (_Float16*)alloc((size_t)MTOK * DD * sizeof(_Float16));
    _Float16* lnh     = (_Float16*)alloc((size_t)MTOK * DD * sizeof(_Float16));
    _Float16* xh      = (_Float16*)alloc((size_t)MTOK * DD * sizeof(_Float16));
    // f16 weight staging
    _Float16* m1w1h = (_Float16*)alloc((size_t)HID * DD * 2);
    _Float16* m1w2h = (_Float16*)alloc((size_t)DD * HID * 2);
    _Float16* m2w1h = (_Float16*)alloc((size_t)HID * DD * 2);
    _Float16* m2w2h = (_Float16*)alloc((size_t)DD * HID * 2);
    _Float16* qkvwh = (_Float16*)alloc((size_t)3 * DD * DD * 2);
    _Float16* owh   = (_Float16*)alloc((size_t)DD * DD * 2);
    _Float16* amw1h = (_Float16*)alloc((size_t)HID * DD * 2);
    _Float16* amw2h = (_Float16*)alloc((size_t)DD * HID * 2);
    _Float16* cwh   = (_Float16*)alloc((size_t)DD * DD * 2);

    // FNO scratch overlaid in buf1 (runs before the MLPs use buf1): 32MB total
    float* tr  = (float*)buf1;       // [B,D,64,16] c: 2 * 2,097,152 f32
    float* ti  = tr + 2097152;
    float* xr  = ti + 2097152;       // [B,D,32,16] c
    float* xi  = xr + 1048576;
    float* ofr = xi + 1048576;       // [B,D,32,16] c
    float* ofi = ofr + 1048576;

    // --- stage f16 operands (once per launch) ---
    cvt_kernel<<<32768, 256, 0, stream>>>(x, xh, (size_t)MTOK * DD);
    cvt_kernel<<<1024, 256, 0, stream>>>(m1w1, m1w1h, (size_t)HID * DD);
    cvt_kernel<<<1024, 256, 0, stream>>>(m1w2, m1w2h, (size_t)DD * HID);
    cvt_kernel<<<1024, 256, 0, stream>>>(m2w1, m2w1h, (size_t)HID * DD);
    cvt_kernel<<<1024, 256, 0, stream>>>(m2w2, m2w2h, (size_t)DD * HID);
    cvt_kernel<<<768, 256, 0, stream>>>(qkvw, qkvwh, (size_t)3 * DD * DD);
    cvt_kernel<<<256, 256, 0, stream>>>(ow, owh, (size_t)DD * DD);
    cvt_kernel<<<1024, 256, 0, stream>>>(amw1, amw1h, (size_t)HID * DD);
    cvt_kernel<<<1024, 256, 0, stream>>>(amw2, amw2h, (size_t)DD * HID);
    cvt_kernel<<<256, 256, 0, stream>>>(cw, cwh, (size_t)DD * DD);

    // --- init + gating ---
    zero_kernel<<<32768, 256, 0, stream>>>(out, (size_t)MTOK * DD);
    mean_concat_kernel<<<32, 256, 0, stream>>>(x, text, feat);
    gating_kernel<<<1, 256, 0, stream>>>(feat, g1w, g1b, g2w, g2b, combine, wout);

    // --- Expert 0: FNO ---
    fno_fft1<<<8192, 256, 0, stream>>>(x, tr, ti);
    fno_fft2<<<4096, 256, 0, stream>>>(tr, ti, xr, xi);
    fno_spectral<<<1024, 256, 0, stream>>>(xr, xi, w1r, w1i, w2r, w2i, ofr, ofi);
    fno_ifft1<<<8192, 256, 0, stream>>>(ofr, ofi, tr, ti);   // reuse tr/ti as u
    gemm_kernel<0, 0, 0, 0><<<1024, 256, 0, stream>>>(
        xh, DD, cwh, DD, cb, buf2, DD, nullptr, 0, nullptr, 0, MTOK, DD, DD);
    fno_final<<<32768, 256, 0, stream>>>(tr, ti, buf2, x, combine, out);

    // --- Expert 1: MLP #1 ---
    gemm_kernel<1, 1, 0, 0><<<4096, 256, 0, stream>>>(
        xh, DD, m1w1h, DD, m1b1, buf1, HID, nullptr, 0, nullptr, 0, MTOK, DD, HID);
    gemm_kernel<0, 0, 1, 1><<<1024, 256, 0, stream>>>(
        (const _Float16*)buf1, HID, m1w2h, HID, m1b2, out, DD, x, DD, combine, 1,
        MTOK, HID, DD);

    // --- Expert 2: window attention ---
    ln_kernel<<<32768, 256, 0, stream>>>(x, ln1w, ln1b, lnh);
    gemm_kernel<0, 1, 0, 0><<<3072, 256, 0, stream>>>(
        lnh, DD, qkvwh, DD, qkvb, buf1, 3 * DD, nullptr, 0, nullptr, 0, MTOK, DD, 3 * DD);
    win_attn_kernel<<<2048, 128, 0, stream>>>((const _Float16*)buf1, buf3);
    gemm_kernel<0, 0, 1, 0><<<1024, 256, 0, stream>>>(
        buf3, DD, owh, DD, ob, buf2, DD, x, DD, nullptr, 0, MTOK, DD, DD);   // v2
    ln_kernel<<<32768, 256, 0, stream>>>(buf2, ln2w, ln2b, lnh);
    gemm_kernel<1, 1, 0, 0><<<4096, 256, 0, stream>>>(
        lnh, DD, amw1h, DD, amb1, buf1, HID, nullptr, 0, nullptr, 0, MTOK, DD, HID);
    gemm_kernel<0, 0, 1, 1><<<1024, 256, 0, stream>>>(
        (const _Float16*)buf1, HID, amw2h, HID, amb2, out, DD, buf2, DD, combine, 2,
        MTOK, HID, DD);

    // --- Expert 3: MLP #2 ---
    gemm_kernel<1, 1, 0, 0><<<4096, 256, 0, stream>>>(
        xh, DD, m2w1h, DD, m2b1, buf1, HID, nullptr, 0, nullptr, 0, MTOK, DD, HID);
    gemm_kernel<0, 0, 1, 1><<<1024, 256, 0, stream>>>(
        (const _Float16*)buf1, HID, m2w2h, HID, m2b2, out, DD, x, DD, combine, 3,
        MTOK, HID, DD);
}
